// _SinkhornHead_88089779241035
// MI455X (gfx1250) — matrix operand order repaired and checked by reading, not compiled, not run on hardware
//
#include <hip/hip_runtime.h>
#include <hip/hip_bf16.h>
#include <math.h>

// ---------------------------------------------------------------------------
// SinkhornHead for MI455X (gfx1250, wave32, WMMA)
//   proj:   H = gelu(X@W1+b1) ; A = H@W2+b2   (bf16 hi/lo split WMMA GEMM)
//   scores: S = (A@B^T) * (1/TEMP)
//   sinkhorn: log_alpha == S + r_i + c_j  -> streaming LSE passes update r,c
//   out = exp(S + r + c)
// ---------------------------------------------------------------------------

typedef __attribute__((ext_vector_type(16))) __bf16 v16bf;
typedef __attribute__((ext_vector_type(8)))  __bf16 v8bf;
typedef __attribute__((ext_vector_type(8)))  float  v8f;

#define BM 128
#define BN 128
#define BK 32
#define LDSP 40   // padded LDS row (bf16 elems): 80B stride -> conflict-free b128

enum { MODE_PROJ_GELU = 0, MODE_PROJ_BIAS = 1, MODE_SCORES = 2 };

__device__ __forceinline__ v16bf combine16(v8bf x, v8bf y) {
  return __builtin_shufflevector(x, y, 0,1,2,3,4,5,6,7,8,9,10,11,12,13,14,15);
}

// A-fragment (16x32 bf16, MxK): lane l: M = l&15, group g = l>>4.
// element e<8 -> K = e + 8g ; e>=8 -> K = e + 8 + 8g  (per ISA VGPR layout)
__device__ __forceinline__ v16bf load_afrag(const __bf16* row, int g) {
  v8bf x = *(const v8bf*)(row + 8 * g);
  v8bf y = *(const v8bf*)(row + 16 + 8 * g);
  return combine16(x, y);
}

// B-fragment (32x16 bf16, KxN): lane l: N = l&15, group g = l>>4.
// element e<8 -> K = e + 8g ; e>=8 -> K = e + 8 + 8g  (matches A-fragment)
__device__ __forceinline__ v16bf load_bfrag(const __bf16* row, int g) {
  v8bf x = *(const v8bf*)(row + 8 * g);
  v8bf y = *(const v8bf*)(row + 16 + 8 * g);
  return combine16(x, y);
}

__device__ __forceinline__ void split_store(float x, __bf16* ph, __bf16* pl) {
  __bf16 h = (__bf16)x;            // RNE truncation to bf16 (hi part)
  *ph = h;
  *pl = (__bf16)(x - (float)h);    // residual (lo part)
}

__device__ __forceinline__ float gelu_exact(float x) {
  return 0.5f * x * (1.0f + erff(x * 0.70710678118654752f));
}

// C[m,n] = sum_k A[m,k] * Bt[n,k]   (Bt conceptually [N,K] row-major)
//  MODE_PROJ_*:  Bg is W [K,N] row-major (ld=N); transposed while filling LDS
//  MODE_SCORES:  Bg is B [N,K] row-major (ld=K); filled directly
template <int MODE>
__global__ __launch_bounds__(256) void gemm_kernel(
    const float* __restrict__ Ag, const float* __restrict__ Bg,
    const float* __restrict__ bias, float* __restrict__ Cg,
    int M, int N, int K) {
  __shared__ __bf16 Ah[BM][LDSP];
  __shared__ __bf16 Al[BM][LDSP];
  __shared__ __bf16 Bh[BN][LDSP];
  __shared__ __bf16 Bl[BN][LDSP];

  const int tid  = threadIdx.x;
  const int lane = tid & 31;
  const int wid  = tid >> 5;     // 8 waves
  const int wm   = wid >> 1;     // 0..3 : 32-row wave stripe
  const int wn   = wid & 1;      // 0..1 : 64-col wave stripe
  const int lg   = lane >> 4;    // lane group (0/1)
  const int ln   = lane & 15;

  const int m0 = blockIdx.y * BM;
  const int n0 = blockIdx.x * BN;

  v8f zero = {};
  v8f acc[2][4];
#pragma unroll
  for (int i = 0; i < 2; ++i)
#pragma unroll
    for (int j = 0; j < 4; ++j) acc[i][j] = zero;

  for (int k0 = 0; k0 < K; k0 += BK) {
    // ---- A tile: BM x BK (f32 -> bf16 hi/lo) ----
#pragma unroll
    for (int it = 0; it < 4; ++it) {
      int idx = tid + 256 * it;  // 0..1023 = 128 rows x 8 float4
      int row = idx >> 3;
      int c4  = idx & 7;
      float4 v = *(const float4*)(Ag + (size_t)(m0 + row) * K + k0 + c4 * 4);
      split_store(v.x, &Ah[row][c4 * 4 + 0], &Al[row][c4 * 4 + 0]);
      split_store(v.y, &Ah[row][c4 * 4 + 1], &Al[row][c4 * 4 + 1]);
      split_store(v.z, &Ah[row][c4 * 4 + 2], &Al[row][c4 * 4 + 2]);
      split_store(v.w, &Ah[row][c4 * 4 + 3], &Al[row][c4 * 4 + 3]);
    }
    // ---- B tile: BN x BK stored N-major as Bs[n][k] ----
    if (MODE == MODE_SCORES) {
#pragma unroll
      for (int it = 0; it < 4; ++it) {
        int idx = tid + 256 * it;
        int row = idx >> 3;
        int c4  = idx & 7;
        float4 v = *(const float4*)(Bg + (size_t)(n0 + row) * K + k0 + c4 * 4);
        split_store(v.x, &Bh[row][c4 * 4 + 0], &Bl[row][c4 * 4 + 0]);
        split_store(v.y, &Bh[row][c4 * 4 + 1], &Bl[row][c4 * 4 + 1]);
        split_store(v.z, &Bh[row][c4 * 4 + 2], &Bl[row][c4 * 4 + 2]);
        split_store(v.w, &Bh[row][c4 * 4 + 3], &Bl[row][c4 * 4 + 3]);
      }
    } else {
      // W rows (coalesced read), transposed scatter into LDS
#pragma unroll
      for (int it = 0; it < 4; ++it) {
        int idx = tid + 256 * it;  // 32 k-rows x 32 float4
        int kr = idx >> 5;
        int c4 = idx & 31;
        float4 v = *(const float4*)(Bg + (size_t)(k0 + kr) * N + n0 + c4 * 4);
        split_store(v.x, &Bh[c4 * 4 + 0][kr], &Bl[c4 * 4 + 0][kr]);
        split_store(v.y, &Bh[c4 * 4 + 1][kr], &Bl[c4 * 4 + 1][kr]);
        split_store(v.z, &Bh[c4 * 4 + 2][kr], &Bl[c4 * 4 + 2][kr]);
        split_store(v.w, &Bh[c4 * 4 + 3][kr], &Bl[c4 * 4 + 3][kr]);
      }
    }
    __syncthreads();

    if (k0 + BK < K) {  // hint next tiles into cache (global_prefetch_b8)
      __builtin_prefetch(Ag + (size_t)(m0 + (tid >> 1)) * K + k0 + BK + (tid & 1) * 16, 0, 1);
      if (MODE == MODE_SCORES)
        __builtin_prefetch(Bg + (size_t)(n0 + (tid >> 1)) * K + k0 + BK + (tid & 1) * 16, 0, 1);
      else
        __builtin_prefetch(Bg + (size_t)(k0 + BK + (tid >> 3)) * N + n0 + (tid & 7) * 16, 0, 1);
    }

    // ---- WMMA compute: 2x4 tiles, 3 products each (hi*hi + hi*lo + lo*hi) ----
    v16bf afh[2], afl[2], bfh[4], bfl[4];
#pragma unroll
    for (int tm = 0; tm < 2; ++tm) {
      int r = wm * 32 + tm * 16 + ln;
      afh[tm] = load_afrag(&Ah[r][0], lg);
      afl[tm] = load_afrag(&Al[r][0], lg);
    }
#pragma unroll
    for (int tn = 0; tn < 4; ++tn) {
      int r = wn * 64 + tn * 16 + ln;
      bfh[tn] = load_bfrag(&Bh[r][0], lg);
      bfl[tn] = load_bfrag(&Bl[r][0], lg);
    }
#pragma unroll
    for (int tm = 0; tm < 2; ++tm)
#pragma unroll
      for (int tn = 0; tn < 4; ++tn) {
        acc[tm][tn] = __builtin_amdgcn_wmma_f32_16x16x32_bf16(
            false, afh[tm], false, bfh[tn], (short)0, acc[tm][tn], false, false);
        acc[tm][tn] = __builtin_amdgcn_wmma_f32_16x16x32_bf16(
            false, afh[tm], false, bfl[tn], (short)0, acc[tm][tn], false, false);
        acc[tm][tn] = __builtin_amdgcn_wmma_f32_16x16x32_bf16(
            false, afl[tm], false, bfh[tn], (short)0, acc[tm][tn], false, false);
      }
    __syncthreads();
  }

  // ---- epilogue: C layout lane l reg r -> (M = r + 8*(l>>4), N = l&15) ----
#pragma unroll
  for (int tm = 0; tm < 2; ++tm)
#pragma unroll
    for (int tn = 0; tn < 4; ++tn)
#pragma unroll
      for (int rr = 0; rr < 8; ++rr) {
        int Mi = m0 + wm * 32 + tm * 16 + rr + 8 * lg;
        int Ni = n0 + wn * 64 + tn * 16 + ln;
        float v = acc[tm][tn][rr];
        if (MODE == MODE_PROJ_GELU) {
          v = gelu_exact(v + bias[Ni]);
        } else if (MODE == MODE_PROJ_BIAS) {
          v = v + bias[Ni];
        } else {
          v = v * 20.0f;  // 1/TEMP
        }
        Cg[(size_t)Mi * N + Ni] = v;
      }
}

// ---------------------------------------------------------------------------
// Sinkhorn streaming kernels: log_alpha == S + r_i + c_j
// ---------------------------------------------------------------------------

__global__ __launch_bounds__(256) void init_rc(float* r, float* c, int N) {
  int j = blockIdx.x * 256 + threadIdx.x;
  if (j < N) { r[j] = 0.0f; c[j] = 0.0f; }
}

// r[i] = -logsumexp_j( S[i][j] + c[j] )  : one wave per row, online LSE
__global__ __launch_bounds__(256) void row_lse(const float* __restrict__ S,
                                               const float* __restrict__ c,
                                               float* __restrict__ r, int N) {
  int row  = blockIdx.x * 8 + (threadIdx.x >> 5);
  int lane = threadIdx.x & 31;
  const float4* Sr = (const float4*)(S + (size_t)row * N);
  const float4* C4 = (const float4*)c;
  float m = -INFINITY, s = 0.0f;
  for (int j = lane; j < (N >> 2); j += 32) {
    float4 v = Sr[j];
    float4 cc = C4[j];
    float x[4] = {v.x + cc.x, v.y + cc.y, v.z + cc.z, v.w + cc.w};
#pragma unroll
    for (int u = 0; u < 4; ++u) {
      float xv = x[u];
      if (xv > m) { s = s * __expf(m - xv) + 1.0f; m = xv; }
      else        { s += __expf(xv - m); }
    }
  }
#pragma unroll
  for (int off = 16; off > 0; off >>= 1) {
    float mo = __shfl_xor(m, off, 32);
    float so = __shfl_xor(s, off, 32);
    float mn = fmaxf(m, mo);
    s = s * __expf(m - mn) + so * __expf(mo - mn);
    m = mn;
  }
  if (lane == 0) r[row] = -(m + __logf(s));
}

// partial column LSE over a stripe of rows (coalesced: thread = column)
__global__ __launch_bounds__(256) void col_partial(const float* __restrict__ S,
                                                   const float* __restrict__ r,
                                                   float* __restrict__ pmax,
                                                   float* __restrict__ psum,
                                                   int N, int rows_per) {
  int j  = blockIdx.x * 256 + threadIdx.x;
  int i0 = blockIdx.y * rows_per;
  float m = -INFINITY, s = 0.0f;
  for (int i = i0; i < i0 + rows_per; ++i) {
    float x = S[(size_t)i * N + j] + r[i];
    if (x > m) { s = s * __expf(m - x) + 1.0f; m = x; }
    else       { s += __expf(x - m); }
  }
  pmax[(size_t)blockIdx.y * N + j] = m;
  psum[(size_t)blockIdx.y * N + j] = s;
}

__global__ __launch_bounds__(256) void col_combine(const float* __restrict__ pmax,
                                                   const float* __restrict__ psum,
                                                   float* __restrict__ c, int N, int R) {
  int j = blockIdx.x * 256 + threadIdx.x;
  float m = -INFINITY, s = 0.0f;
  for (int k = 0; k < R; ++k) {
    float mk = pmax[(size_t)k * N + j];
    float sk = psum[(size_t)k * N + j];
    float mn = fmaxf(m, mk);
    s = s * __expf(m - mn) + sk * __expf(mk - mn);
    m = mn;
  }
  c[j] = -(m + __logf(s));
}

__global__ __launch_bounds__(256) void final_exp(const float* __restrict__ S,
                                                 const float* __restrict__ r,
                                                 const float* __restrict__ c,
                                                 float* __restrict__ out, int N) {
  size_t idx4 = (size_t)blockIdx.x * 256 + threadIdx.x;  // float4 index
  int nq = N >> 2;
  int row = (int)(idx4 / nq);
  int c4  = (int)(idx4 % nq);
  float4 v  = ((const float4*)(S + (size_t)row * N))[c4];
  float4 cc = ((const float4*)c)[c4];
  float rv = r[row];
  float4 o;
  o.x = __expf(v.x + rv + cc.x);
  o.y = __expf(v.y + rv + cc.y);
  o.z = __expf(v.z + rv + cc.z);
  o.w = __expf(v.w + rv + cc.w);
  ((float4*)(out + (size_t)row * N))[c4] = o;
}

// ---------------------------------------------------------------------------

extern "C" void kernel_launch(void* const* d_in, const int* in_sizes, int n_in,
                              void* d_out, int out_size, void* d_ws, size_t ws_size,
                              hipStream_t stream) {
  const float* emb_a = (const float*)d_in[0];
  const float* emb_b = (const float*)d_in[1];
  const float* W1    = (const float*)d_in[2];
  const float* b1    = (const float*)d_in[3];
  const float* W2    = (const float*)d_in[4];
  const float* b2    = (const float*)d_in[5];
  float* out = (float*)d_out;

  const int N = 8192, D = 1024, ITERS = 20, RSPLIT = 32;

  float* ws   = (float*)d_ws;
  float* S    = ws;                        // N*N      (256 MB)
  float* Abuf = ws + (size_t)N * N;        // N*D
  float* Bbuf = Abuf + (size_t)N * D;      // N*D
  float* r    = Bbuf + (size_t)N * D;      // N
  float* c    = r + N;                     // N
  float* pmax = c + N;                     // RSPLIT*N
  float* psum = pmax + (size_t)RSPLIT * N; // RSPLIT*N
  float* H    = S;  // hidden activations alias S (used strictly before S)

  dim3 blk(256);
  dim3 gproj(D / BN, N / BM);  // (8, 64)
  dim3 gsc(N / BN, N / BM);    // (64, 64)

  // Projections (H aliases S; S is written only after H's last use)
  gemm_kernel<MODE_PROJ_GELU><<<gproj, blk, 0, stream>>>(emb_a, W1, b1, H, N, D, D);
  gemm_kernel<MODE_PROJ_BIAS><<<gproj, blk, 0, stream>>>(H, W2, b2, Abuf, N, D, D);
  gemm_kernel<MODE_PROJ_GELU><<<gproj, blk, 0, stream>>>(emb_b, W1, b1, H, N, D, D);
  gemm_kernel<MODE_PROJ_BIAS><<<gproj, blk, 0, stream>>>(H, W2, b2, Bbuf, N, D, D);

  // Scores: S = (A @ B^T) / TEMP
  gemm_kernel<MODE_SCORES><<<gsc, blk, 0, stream>>>(Abuf, Bbuf, nullptr, S, N, N, D);

  // Sinkhorn iterations on (r, c) only — S is read-only streaming
  init_rc<<<dim3(N / 256), blk, 0, stream>>>(r, c, N);
  for (int it = 0; it < ITERS; ++it) {
    row_lse<<<dim3(N / 8), blk, 0, stream>>>(S, c, r, N);
    col_partial<<<dim3(N / 256, RSPLIT), blk, 0, stream>>>(S, r, pmax, psum, N, N / RSPLIT);
    col_combine<<<dim3(N / 256), blk, 0, stream>>>(pmax, psum, c, N, RSPLIT);
  }

  // out = exp(S + r + c)
  final_exp<<<dim3((unsigned)((size_t)N * N / 4 / 256)), blk, 0, stream>>>(S, r, c, out, N);
}
